// TritonSelfAttention_70798240907513
// MI455X (gfx1250) — compile-verified, matrix operand order checked by reading, not run on hardware
//
#include <hip/hip_runtime.h>

typedef __bf16 bf16;
typedef __attribute__((ext_vector_type(4)))  __bf16 v4bf;
typedef __attribute__((ext_vector_type(8)))  __bf16 v8bf;
typedef __attribute__((ext_vector_type(16))) __bf16 v16bf;
typedef __attribute__((ext_vector_type(8)))  float  v8f;
typedef __attribute__((ext_vector_type(4)))  unsigned int u32x4;
typedef __attribute__((ext_vector_type(4)))  int i32x4;
typedef __attribute__((ext_vector_type(8)))  int i32x8;

#define AS1 __attribute__((address_space(1)))
#define AS3 __attribute__((address_space(3)))

#define S_LEN 2048
#define HID   4096
#define HEADS 32
#define HD    128
#define NQKV  (3 * HID)          // 12288
#define QKV_LD NQKV
#define ATTN_SCALE 0.08838834764831845f   // 1/sqrt(128)

// Feature probes (fall back to the proven synchronous path if absent).
#if __has_builtin(__builtin_amdgcn_global_load_async_to_lds_b128)
#define HAVE_ASYNC_LDS 1
#else
#define HAVE_ASYNC_LDS 0
#endif
#if __has_builtin(__builtin_amdgcn_tensor_load_to_lds)
#define HAVE_TDM 1
#else
#define HAVE_TDM 0
#endif

// ---------------------------------------------------------------------------
// CDNA5 primitives
// ---------------------------------------------------------------------------
__device__ __forceinline__ v8f wmma_bf16f32(v16bf a, v16bf b, v8f c) {
  return __builtin_amdgcn_wmma_f32_16x16x32_bf16(
      /*neg_a=*/false, a, /*neg_b=*/false, b,
      /*c_mod=*/(short)0, c, /*reuse_a=*/false, /*reuse_b=*/false);
}

// 16-byte global -> LDS copy: async DMA (ASYNCcnt) when available.
// Builtin prototype (from compiler diagnostic): (v4i AS1*, v4i AS3*, Ii, Ii).
__device__ __forceinline__ void async_cp16(const bf16* g, bf16* l) {
#if HAVE_ASYNC_LDS
  __builtin_amdgcn_global_load_async_to_lds_b128(
      (AS1 i32x4*)g, (AS3 i32x4*)l, /*offset=*/0, /*cpol=*/0);
#else
  *(v8bf*)l = *(const v8bf*)g;
#endif
}

__device__ __forceinline__ void wait_async0() {
#if HAVE_ASYNC_LDS
#if __has_builtin(__builtin_amdgcn_s_wait_asynccnt)
  __builtin_amdgcn_s_wait_asynccnt(0);
#else
  asm volatile("s_wait_asynccnt 0x0" ::: "memory");
#endif
#endif
}

// Fragment slice per the CDNA5 16-bit A/B layout:
// lane<16 -> K {k..k+7, k+16..k+23}; lane>=16 caller passes k+8.
__device__ __forceinline__ v16bf load_frag(const bf16* base, int ld, int row, int k) {
  const bf16* p = base + row * ld + k;
  v8bf lo = *(const v8bf*)p;
  v8bf hi = *(const v8bf*)(p + 16);
  v16bf r;
#pragma unroll
  for (int i = 0; i < 8; ++i) { r[i] = lo[i]; r[i + 8] = hi[i]; }
  return r;
}

// ---------------------------------------------------------------------------
// fp32 -> bf16 conversion (vectorized)
// ---------------------------------------------------------------------------
__global__ __launch_bounds__(256) void cvt_f32_to_bf16(
    const float* __restrict__ in, bf16* __restrict__ out, int n) {
  int i = (blockIdx.x * 256 + threadIdx.x) * 4;
  if (i + 4 <= n) {
    float4 v = *(const float4*)(in + i);
    v4bf o;
    o[0] = (bf16)v.x; o[1] = (bf16)v.y; o[2] = (bf16)v.z; o[3] = (bf16)v.w;
    *(v4bf*)(out + i) = o;
  } else {
    for (; i < n; ++i) out[i] = (bf16)in[i];
  }
}

// ---------------------------------------------------------------------------
// Tiled GEMM: C[M,N] = A[M,K](bf16,row) * B[K,N](bf16,row) + bias[N]
// 128x128 block tile, K-step 32, double-buffered LDS, async A-tile copies.
// 8 waves (wave32): 4 along M x 2 along N, each wave 32x64 = 2x4 WMMA tiles.
// ---------------------------------------------------------------------------
template <typename OutT>
__global__ __launch_bounds__(256) void gemm_bf16_tn(
    const bf16* __restrict__ A, const bf16* __restrict__ B,
    const float* __restrict__ bias, OutT* __restrict__ C,
    int M, int N, int K) {
  __shared__ bf16 As[2][128 * 40];   // [m][k], padded stride 40
  __shared__ bf16 Bs[2][128 * 40];   // [n][k] (transposed), padded stride 40

  const int tid  = threadIdx.x;
  const int lane = tid & 31;
  const int wave = tid >> 5;
  const int wm   = (wave & 3) * 32;
  const int wn   = (wave >> 2) * 64;
  const int lr   = lane & 15;
  const int klo  = (lane < 16) ? 0 : 8;
  const int m0   = blockIdx.y * 128;
  const int n0   = blockIdx.x * 128;

  v8f acc[2][4] = {};

  auto load_tiles = [&](int buf, int kt) {
    // A tile 128x32: contiguous 16B chunks in both global and LDS -> async DMA
#pragma unroll
    for (int c = tid; c < 512; c += 256) {
      int row = c >> 2, kc = (c & 3) * 8;
      async_cp16(A + (size_t)(m0 + row) * K + kt + kc,
                 &As[buf][row * 40 + kc]);
    }
    // B tile 32x128 loaded coalesced, stored transposed [n][k]
#pragma unroll
    for (int c = tid; c < 512; c += 256) {
      int kk = c >> 4, nc = (c & 15) * 8;
      v8bf v = *(const v8bf*)(B + (size_t)(kt + kk) * N + n0 + nc);
#pragma unroll
      for (int i = 0; i < 8; ++i) Bs[buf][(nc + i) * 40 + kk] = v[i];
    }
  };

  load_tiles(0, 0);
  wait_async0();
  __syncthreads();

  int buf = 0;
  for (int kt = 0; kt < K; kt += 32) {
    if (kt + 32 < K) load_tiles(buf ^ 1, kt + 32);  // prefetch next tile

    v16bf af[2], bfr[4];
#pragma unroll
    for (int mt = 0; mt < 2; ++mt)
      af[mt] = load_frag(&As[buf][0], 40, wm + mt * 16 + lr, klo);
#pragma unroll
    for (int nt = 0; nt < 4; ++nt)
      bfr[nt] = load_frag(&Bs[buf][0], 40, wn + nt * 16 + lr, klo);
#pragma unroll
    for (int mt = 0; mt < 2; ++mt)
#pragma unroll
      for (int nt = 0; nt < 4; ++nt)
        acc[mt][nt] = wmma_bf16f32(af[mt], bfr[nt], acc[mt][nt]);

    wait_async0();     // next buffer's async copies landed
    __syncthreads();   // single barrier per k-step (double buffering)
    buf ^= 1;
  }

  // Epilogue: D layout -> VGPR j holds row (j + 8*(lane>=16)), col = lane%16
  const int rb = (lane < 16) ? 0 : 8;
#pragma unroll
  for (int mt = 0; mt < 2; ++mt)
#pragma unroll
    for (int nt = 0; nt < 4; ++nt) {
      int col = n0 + wn + nt * 16 + lr;
      float bv = bias[col];
#pragma unroll
      for (int j = 0; j < 8; ++j) {
        int row = m0 + wm + mt * 16 + rb + j;
        C[(size_t)row * N + col] = (OutT)(acc[mt][nt][j] + bv);
      }
    }
}

// ---------------------------------------------------------------------------
// Flash attention, causal. Block = (head, 128 query rows), 8 waves; each
// wave owns 16 query rows. KV tiles of 64 keys. Online softmax in f32.
// Q tile is fetched by the Tensor Data Mover; K tiles by async-to-LDS.
// ---------------------------------------------------------------------------
__global__ __launch_bounds__(256) void flash_attn_kernel(
    const bf16* __restrict__ qkv, bf16* __restrict__ ctx) {
  __shared__ bf16 Qs[128 * 136];   // [q][d], stride 136 (TDM pad reproduces it)
  __shared__ bf16 Ks[64 * 136];    // [key][d]  (transposed-B for Q*K^T)
  __shared__ bf16 Vs[128 * 72];    // [d][key]  (transposed-B for P*V)
  __shared__ bf16 Ps[8 * 16 * 72]; // per-wave P staging (D-layout -> A-layout)

  const int h    = blockIdx.x;
  const int q0   = blockIdx.y * 128;
  const int tid  = threadIdx.x;
  const int lane = tid & 31;
  const int wave = tid >> 5;
  const int lr   = lane & 15;
  const int klo  = (lane < 16) ? 0 : 8;
  const int rb   = (lane < 16) ? 0 : 8;
  const int wq   = wave * 16;

  const bf16* Qg = qkv + (size_t)q0 * QKV_LD + h * HD;
  const bf16* Kg = qkv + HID + h * HD;
  const bf16* Vg = qkv + 2 * HID + h * HD;

  // ---- Q tile -> LDS --------------------------------------------------
#if HAVE_TDM
  if (wave == 0) {
    // D# group0: count=1 | lds_addr | global_addr | type=2 ("image")
    unsigned long long ga = (unsigned long long)(size_t)Qg;
    unsigned lds_base = (unsigned)(AS3 bf16*)Qs;
    u32x4 g0;
    g0[0] = 1u;                                   // count=1, user mode
    g0[1] = lds_base;                             // lds_addr (bytes)
    g0[2] = (unsigned)(ga & 0xffffffffu);         // global_addr[31:0]
    g0[3] = (unsigned)((ga >> 32) & 0x01ffffffu)  // global_addr[56:32]
            | (2u << 30);                         // type=2
    // D# group1: data_size=2B, pad 4 dwords per 64 dwords (stride 136 bf16),
    // tensor 12288 x 2048, tile 128 x 128, dim0 stride 12288.
    i32x8 g1 = {};
    g1[0] = (int)((1u << 16)      // data_size = 2 bytes
                  | (1u << 20)    // pad_enable
                  | (5u << 22)    // pad_interval: 2^(5+1)=64 dwords
                  | (3u << 25));  // pad_amount: 3+1 = 4 dwords
    g1[1] = (int)((12288u & 0xffffu) << 16);            // tensor_dim0 lo
    g1[2] = (int)((12288u >> 16) | ((2048u & 0xffffu) << 16)); // dim0 hi|dim1 lo
    g1[3] = (int)((2048u >> 16) | (128u << 16));        // dim1 hi | tile_dim0
    g1[4] = 128;                                        // tile_dim1
    g1[5] = 12288;                                      // tensor_dim0_stride lo
    i32x4 z4 = {};
    i32x8 z8 = {};
#if defined(__clang_major__) && (__clang_major__ >= 23)
    __builtin_amdgcn_tensor_load_to_lds(g0, g1, z4, z4, z8, 0);
#else
    __builtin_amdgcn_tensor_load_to_lds(g0, g1, z4, z4, 0);
#endif
    __builtin_amdgcn_s_wait_tensorcnt(0);
  }
#else
  for (int c = tid; c < 2048; c += 256) {
    int row = c >> 4, dc = (c & 15) * 8;
    *(v8bf*)(Qs + row * 136 + dc) =
        *(const v8bf*)(Qg + (size_t)row * QKV_LD + dc);
  }
#endif
  __syncthreads();

  // Q fragments live in registers for the whole KV loop (4 x v16bf)
  v16bf qf[4];
#pragma unroll
  for (int d = 0; d < 4; ++d)
    qf[d] = load_frag(Qs, 136, wq + lr, d * 32 + klo);

  float mrow[8], lsum[8];
  v8f oacc[8] = {};
#pragma unroll
  for (int j = 0; j < 8; ++j) { mrow[j] = -3.0e38f; lsum[j] = 0.f; }

  bf16* Pw = Ps + wave * (16 * 72);
  const int kv_end = q0 + 128;

  for (int kv0 = 0; kv0 < kv_end; kv0 += 64) {
    __syncthreads();   // previous iter's reads of Ks/Vs complete
    // K tile 64x128: contiguous rows -> async DMA to LDS
    for (int c = tid; c < 1024; c += 256) {
      int row = c >> 4, dc = (c & 15) * 8;
      async_cp16(Kg + (size_t)(kv0 + row) * QKV_LD + dc,
                 Ks + row * 136 + dc);
    }
    // V tile transposed: Vs[d][key] (scatter; VGPR path)
    for (int c = tid; c < 1024; c += 256) {
      int key = c >> 4, dc = (c & 15) * 8;
      v8bf v = *(const v8bf*)(Vg + (size_t)(kv0 + key) * QKV_LD + dc);
#pragma unroll
      for (int i = 0; i < 8; ++i) Vs[(dc + i) * 72 + key] = v[i];
    }
    wait_async0();
    __syncthreads();

    // S = Q * K^T  : 4 d-steps x 4 key-tiles
    v8f sacc[4] = {};
#pragma unroll
    for (int d = 0; d < 4; ++d) {
#pragma unroll
      for (int nt = 0; nt < 4; ++nt) {
        v16bf kf = load_frag(Ks, 136, nt * 16 + lr, d * 32 + klo);
        sacc[nt] = wmma_bf16f32(qf[d], kf, sacc[nt]);
      }
    }

    // Online softmax. D layout: lane holds col=lane%16, rows rb+j.
#pragma unroll
    for (int j = 0; j < 8; ++j) {
      const int qrow = q0 + wq + rb + j;
      float mx = mrow[j];
#pragma unroll
      for (int nt = 0; nt < 4; ++nt) {
        int key = kv0 + nt * 16 + lr;
        float s = sacc[nt][j] * ATTN_SCALE;
        s = (key <= qrow) ? s : -3.0e38f;   // causal mask
        sacc[nt][j] = s;
        mx = fmaxf(mx, s);
      }
#pragma unroll
      for (int off = 8; off >= 1; off >>= 1)
        mx = fmaxf(mx, __shfl_xor(mx, off, 32));
      float corr = __expf(mrow[j] - mx);
      mrow[j] = mx;
      lsum[j] *= corr;
#pragma unroll
      for (int d8 = 0; d8 < 8; ++d8) oacc[d8][j] *= corr;
      float rs = 0.f;
#pragma unroll
      for (int nt = 0; nt < 4; ++nt) {
        float e = __expf(sacc[nt][j] - mx);
        sacc[nt][j] = e;
        rs += e;
      }
#pragma unroll
      for (int off = 8; off >= 1; off >>= 1)
        rs += __shfl_xor(rs, off, 32);
      lsum[j] += rs;
      // stage P (bf16) in per-wave LDS: D-layout -> A-matrix layout
#pragma unroll
      for (int nt = 0; nt < 4; ++nt)
        Pw[(rb + j) * 72 + nt * 16 + lr] = (bf16)sacc[nt][j];
    }
    // same-wave LDS RAW across lanes: drain DS counter explicitly
    asm volatile("s_wait_dscnt 0x0" ::: "memory");

    // O += P * V : 2 key-steps x 8 d-tiles
#pragma unroll
    for (int ks = 0; ks < 2; ++ks) {
      v16bf pf = load_frag(Pw, 72, lr, ks * 32 + klo);
#pragma unroll
      for (int d8 = 0; d8 < 8; ++d8) {
        v16bf vf = load_frag(Vs, 72, d8 * 16 + lr, ks * 32 + klo);
        oacc[d8] = wmma_bf16f32(pf, vf, oacc[d8]);
      }
    }
  }

  // Normalize and write ctx[s][h*128 + d] (bf16)
  bf16* Cg = ctx + (size_t)(q0 + wq) * HID + h * HD;
#pragma unroll
  for (int j = 0; j < 8; ++j) {
    float inv = 1.0f / lsum[j];
    int r = rb + j;
#pragma unroll
    for (int d8 = 0; d8 < 8; ++d8)
      Cg[(size_t)r * HID + d8 * 16 + lr] = (bf16)(oacc[d8][j] * inv);
  }
}

// ---------------------------------------------------------------------------
// Host-side launch
// ---------------------------------------------------------------------------
extern "C" void kernel_launch(void* const* d_in, const int* in_sizes, int n_in,
                              void* d_out, int out_size, void* d_ws, size_t ws_size,
                              hipStream_t stream) {
  (void)in_sizes; (void)n_in; (void)out_size; (void)ws_size;
  const float* x    = (const float*)d_in[0];  // [2048,4096]
  const float* qkvw = (const float*)d_in[1];  // [4096,12288]
  const float* qkvb = (const float*)d_in[2];  // [12288]
  const float* ow   = (const float*)d_in[3];  // [4096,4096]
  const float* ob   = (const float*)d_in[4];  // [4096]
  float* out = (float*)d_out;                 // [2048,4096]

  // Workspace layout (bf16)
  bf16* xb    = (bf16*)d_ws;                        // 2048*4096
  bf16* wqkvb = xb    + (size_t)S_LEN * HID;        // 4096*12288
  bf16* wob   = wqkvb + (size_t)HID * NQKV;         // 4096*4096
  bf16* qkv   = wob   + (size_t)HID * HID;          // 2048*12288
  bf16* ctx   = qkv   + (size_t)S_LEN * NQKV;       // 2048*4096

  auto cvt = [&](const float* src, bf16* dst, int n) {
    int blocks = (n / 4 + 255) / 256;
    cvt_f32_to_bf16<<<blocks, 256, 0, stream>>>(src, dst, n);
  };
  cvt(x,    xb,    S_LEN * HID);
  cvt(qkvw, wqkvb, HID * NQKV);
  cvt(ow,   wob,   HID * HID);

  // 1) QKV projection -> bf16 qkv
  dim3 g1(NQKV / 128, S_LEN / 128);
  gemm_bf16_tn<bf16><<<g1, dim3(256), 0, stream>>>(
      xb, wqkvb, qkvb, qkv, S_LEN, NQKV, HID);

  // 2) causal flash attention -> bf16 ctx
  dim3 g2(HEADS, S_LEN / 128);
  flash_attn_kernel<<<g2, dim3(256), 0, stream>>>(qkv, ctx);

  // 3) output projection -> fp32 out
  dim3 g3(HID / 128, S_LEN / 128);
  gemm_bf16_tn<float><<<g3, dim3(256), 0, stream>>>(
      ctx, wob, ob, out, S_LEN, HID, HID);
}
